// RecurrentSlotAttention_61375082660411
// MI455X (gfx1250) — compile-verified, matrix operand order checked by reading, not run on hardware
//
#include <hip/hip_runtime.h>
#include <hip/hip_bf16.h>
#include <math.h>

typedef __bf16 bf16;
typedef __attribute__((ext_vector_type(16))) __bf16 v16bf;
typedef __attribute__((ext_vector_type(8)))  float  v8f;
typedef __attribute__((ext_vector_type(4)))  unsigned v4u;
typedef __attribute__((ext_vector_type(4)))  int      v4i;
typedef __attribute__((ext_vector_type(8)))  int      v8i;

#define BB 16
#define NN 4096
#define KK 16
#define DF 1024
#define DD 768
#define HH 12
#define HD 64

#if defined(__has_builtin)
#if __has_builtin(__builtin_amdgcn_tensor_load_to_lds) && __has_builtin(__builtin_amdgcn_s_wait_tensorcnt)
#define HAVE_TDM 1
#endif
#endif

static __device__ __forceinline__ v8f wmma_bf16(v16bf a, v16bf b, v8f c) {
  return __builtin_amdgcn_wmma_f32_16x16x32_bf16(false, a, false, b, (short)0, c, false, false);
}

// ---------------------------------------------------------------------------
// Kernel 0: fp32 -> bf16 weight conversion (one-time)
// ---------------------------------------------------------------------------
__global__ void cvt_bf16_kernel(const float* __restrict__ src, bf16* __restrict__ dst, int n) {
  int i = blockIdx.x * 256 + threadIdx.x;
  if (i < n) dst[i] = (bf16)src[i];
}

// ---------------------------------------------------------------------------
// Kernel 1: per-row LN stats for features (rows = B*N, cdim = DF); wave/row
// ---------------------------------------------------------------------------
__global__ void stats_kernel(const float* __restrict__ x, float* __restrict__ meanA,
                             float* __restrict__ rstdA, int rows, int cdim) {
  int w = threadIdx.x >> 5, lane = threadIdx.x & 31;
  int row = blockIdx.x * 8 + w;
  if (row >= rows) return;
  const float* xp = x + (size_t)row * cdim;
  float s = 0.f, sq = 0.f;
  for (int c = lane; c < cdim; c += 32) { float v = xp[c]; s += v; sq += v * v; }
#pragma unroll
  for (int m = 1; m < 32; m <<= 1) { s += __shfl_xor(s, m, 32); sq += __shfl_xor(sq, m, 32); }
  if (lane == 0) {
    float mean = s / (float)cdim;
    float var = sq / (float)cdim - mean * mean;
    meanA[row] = mean;
    rstdA[row] = rsqrtf(var + 1e-5f);
  }
}

// ---------------------------------------------------------------------------
// Kernel 2: fused LN(features) -> fp GEMM (LDS) -> k GEMM + RoPE, v GEMM
// grid = (B*N)/16 blocks of 256 threads (8 waves); block owns 16 rows,
// wave w owns output columns [96w, 96w+96) = 6 WMMA n-tiles. bf16 weights.
// ---------------------------------------------------------------------------
__global__ void __launch_bounds__(256)
feat_kv_kernel(const float* __restrict__ feat, const int* __restrict__ fpos,
               const float* __restrict__ lnw, const float* __restrict__ lnb,
               const bf16* __restrict__ Wfp, const float* __restrict__ bfp,
               const bf16* __restrict__ Wk, const float* __restrict__ bk,
               const bf16* __restrict__ Wv, const float* __restrict__ bv,
               bf16* __restrict__ krope, bf16* __restrict__ vout,
               const float* __restrict__ meanA, const float* __restrict__ rstdA) {
  __shared__ __align__(64) bf16 s_fp[16][DD];     // 24 KB
  __shared__ float s_mean[16], s_rstd[16];

  int R0 = blockIdx.x * 16;            // global row base (row = b*N + n)
  int b  = R0 >> 12;
  int n0 = R0 & (NN - 1);
  int tid = threadIdx.x;

  if (tid < 16) { s_mean[tid] = meanA[R0 + tid]; s_rstd[tid] = rstdA[R0 + tid]; }
  __syncthreads();

  int w = tid >> 5, lane = tid & 31;
  int arow = lane & 15;
  int chalf = (lane >> 4) << 4;
  int colb = w * 96;

  // ---- phase A: fp tile = LN(features) @ Wfp^T  (contraction DF=1024) ----
  v8f acc[6];
#pragma unroll
  for (int t = 0; t < 6; ++t) acc[t] = (v8f)(0.f);

  for (int c = 0; c < DF; c += 32) {
    const float* fr = feat + (size_t)(R0 + arow) * DF + c + chalf;
    float mu = s_mean[arow], rs = s_rstd[arow];
    v16bf a;
#pragma unroll
    for (int i = 0; i < 16; ++i) {
      float x = (fr[i] - mu) * rs;
      x = x * lnw[c + chalf + i] + lnb[c + chalf + i];
      a[i] = (bf16)x;
    }
#pragma unroll
    for (int t = 0; t < 6; ++t) {
      int col = colb + t * 16 + (lane & 15);
      v16bf bm = *(const v16bf*)(Wfp + (size_t)col * DF + c + chalf);
      acc[t] = wmma_bf16(a, bm, acc[t]);
    }
  }
#pragma unroll
  for (int t = 0; t < 6; ++t) {
    int col = colb + t * 16 + (lane & 15);
#pragma unroll
    for (int r = 0; r < 8; ++r) {
      int rowc = r + ((lane >> 4) << 3);
      s_fp[rowc][col] = (bf16)(acc[t][r] + bfp[col]);
    }
  }
  __syncthreads();

  // ---- phase B: k = fp @ Wk^T + bk, then RoPE2D, store bf16 ----
#pragma unroll
  for (int t = 0; t < 6; ++t) acc[t] = (v8f)(0.f);
  for (int c = 0; c < DD; c += 32) {
    v16bf a = *(const v16bf*)&s_fp[arow][c + chalf];
#pragma unroll
    for (int t = 0; t < 6; ++t) {
      int col = colb + t * 16 + (lane & 15);
      v16bf bm = *(const v16bf*)(Wk + (size_t)col * DD + c + chalf);
      acc[t] = wmma_bf16(a, bm, acc[t]);
    }
  }
#pragma unroll
  for (int p = 0; p < 3; ++p) {
    int col_lo = colb + 32 * p + (lane & 15);
    int hd = col_lo & 63;
    int sel = (hd < 32) ? 0 : 1;
    int j = lane & 15;
    float inv = __expf(-(float)(2 * j) * (1.0f / 32.0f) * 4.6051702f); // 100^(-2j/32)
    float blo = bk[col_lo], bhi = bk[col_lo + 16];
#pragma unroll
    for (int r = 0; r < 8; ++r) {
      int rowc = r + ((lane >> 4) << 3);
      int n = n0 + rowc;
      float pv = (float)fpos[((size_t)b * NN + n) * 2 + sel];
      float ang = pv * inv;
      float co = __cosf(ang), si = __sinf(ang);
      float vlo = acc[2 * p][r] + blo;
      float vhi = acc[2 * p + 1][r] + bhi;
      size_t base = ((size_t)b * NN + n) * DD;
      krope[base + col_lo]      = (bf16)(vlo * co - vhi * si);
      krope[base + col_lo + 16] = (bf16)(vhi * co + vlo * si);
    }
  }

  // ---- phase C: v = fp @ Wv^T + bv, store bf16 ----
#pragma unroll
  for (int t = 0; t < 6; ++t) acc[t] = (v8f)(0.f);
  for (int c = 0; c < DD; c += 32) {
    v16bf a = *(const v16bf*)&s_fp[arow][c + chalf];
#pragma unroll
    for (int t = 0; t < 6; ++t) {
      int col = colb + t * 16 + (lane & 15);
      v16bf bm = *(const v16bf*)(Wv + (size_t)col * DD + c + chalf);
      acc[t] = wmma_bf16(a, bm, acc[t]);
    }
  }
#pragma unroll
  for (int t = 0; t < 6; ++t) {
    int col = colb + t * 16 + (lane & 15);
    float bia = bv[col];
#pragma unroll
    for (int r = 0; r < 8; ++r) {
      int rowc = r + ((lane >> 4) << 3);
      vout[((size_t)b * NN + n0 + rowc) * DD + col] = (bf16)(acc[t][r] + bia);
    }
  }
}

// ---------------------------------------------------------------------------
// Kernel 3: LN over rows; one wave per row
// ---------------------------------------------------------------------------
__global__ void ln_rows_kernel(const float* __restrict__ x, const float* __restrict__ wt,
                               const float* __restrict__ bs, float* __restrict__ out,
                               int rows, int cdim) {
  int w = threadIdx.x >> 5, lane = threadIdx.x & 31;
  int row = blockIdx.x * 8 + w;
  if (row >= rows) return;
  const float* xp = x + (size_t)row * cdim;
  float s = 0.f, sq = 0.f;
  for (int c = lane; c < cdim; c += 32) { float v = xp[c]; s += v; sq += v * v; }
#pragma unroll
  for (int m = 1; m < 32; m <<= 1) { s += __shfl_xor(s, m, 32); sq += __shfl_xor(sq, m, 32); }
  float mean = s / (float)cdim;
  float rstd = rsqrtf(sq / (float)cdim - mean * mean + 1e-5f);
  float* op = out + (size_t)row * cdim;
  for (int c = lane; c < cdim; c += 32) op[c] = (xp[c] - mean) * rstd * wt[c] + bs[c];
}

// ---------------------------------------------------------------------------
// Kernel 4: generic matvec: out[r,d] = act(dot(in[r,:], W[d,:]) + bias[d]) (+res)
// ---------------------------------------------------------------------------
__global__ void matvec_kernel(const float* __restrict__ in, const float* __restrict__ W,
                              const float* __restrict__ bias, const float* __restrict__ res,
                              float* __restrict__ out, int rows, int cols, int cdim, int act) {
  int idx = blockIdx.x * 256 + threadIdx.x;
  if (idx >= rows * cols) return;
  int r = idx / cols, d = idx - r * cols;
  const float* ip = in + (size_t)r * cdim;
  const float* wp = W + (size_t)d * cdim;
  float a = bias[d];
  for (int c = 0; c < cdim; ++c) a += ip[c] * wp[c];
  if (act == 1) a = fmaxf(a, 0.f);
  else if (act == 2) a = 1.f / (1.f + __expf(-a));
  if (res) a += res[idx];
  out[idx] = a;
}

// ---------------------------------------------------------------------------
// Kernel 5: q RoPE (2D, slot positions) -> bf16 (B,H,K,HD)
// ---------------------------------------------------------------------------
__global__ void qrope_kernel(const float* __restrict__ qraw, bf16* __restrict__ qbf) {
  int idx = blockIdx.x * 256 + threadIdx.x;
  if (idx >= BB * HH * KK * HD) return;
  int hd = idx & 63;
  int k = (idx >> 6) & 15;
  int h = (idx >> 10) % HH;
  int b = idx / (HH * KK * HD);
  size_t base = ((size_t)b * KK + k) * DD + h * HD;
  float val = qraw[base + hd];
  int i32 = hd & 31;
  int sel = hd >> 5;
  int j = i32 & 15;
  int phd = (i32 < 16) ? hd + 16 : hd - 16;
  float pv = qraw[base + phd];
  float p = (sel == 0) ? (float)(k >> 2) : (float)(k & 3); // slot pos (y, x), width=4
  float inv = __expf(-(float)(2 * j) * (1.0f / 32.0f) * 4.6051702f);
  float ang = p * inv;
  float co = __cosf(ang), si = __sinf(ang);
  float rot = (i32 < 16) ? -pv : pv;
  qbf[idx] = (bf16)(val * co + rot * si);
}

// ---------------------------------------------------------------------------
// Kernel 6: flash attention. grid = B*H blocks of 256 threads (8 waves).
// Wave w handles n in [w*512,(w+1)*512). V tiles staged into LDS via the
// Tensor Data Mover (D# descriptor, tensor_load_to_lds + s_wait_tensorcnt);
// per-lane vector-load fallback if the builtin is unavailable.
// ---------------------------------------------------------------------------
__global__ void __launch_bounds__(256)
attn_kernel(const bf16* __restrict__ q, const bf16* __restrict__ kr,
            const bf16* __restrict__ vv, float* __restrict__ upd) {
  __shared__ __align__(64) unsigned char smem[40960];
  int b = blockIdx.x / HH, h = blockIdx.x % HH;
  int w = threadIdx.x >> 5, lane = threadIdx.x & 31;
  int chalf = (lane >> 4) << 4;
  int col = lane & 15;

  bf16* lp = (bf16*)smem + w * 16 * 32;                 // [16][32] P tile
  bf16* lv = (bf16*)(smem + 8192) + w * 32 * 64;        // [32][64] V tile

  const bf16* qp = q + ((size_t)(b * HH + h) * KK + (lane & 15)) * HD;
  v16bf aq0 = *(const v16bf*)(qp + chalf);
  v16bf aq1 = *(const v16bf*)(qp + 32 + chalf);

  float m[8], l[8];
  v8f acc[4];
#pragma unroll
  for (int r = 0; r < 8; ++r) { m[r] = -1e30f; l[r] = 0.f; }
#pragma unroll
  for (int d = 0; d < 4; ++d) acc[d] = (v8f)(0.f);

#ifdef HAVE_TDM
  unsigned lds_addr = (unsigned)(uintptr_t)lv;          // low 32 bits = LDS byte offset
#endif

  for (int nt = 0; nt < 16; ++nt) {
    int n = w * 512 + nt * 32;

#ifdef HAVE_TDM
    { // TDM: 2-D tile, 64 x 32 elements of 2 bytes, row stride 768 elems
      unsigned long long ga =
          (unsigned long long)(uintptr_t)(vv + ((size_t)b * NN + n) * DD + h * HD);
      v4u g0;
      g0[0] = 1u;                                   // count=1, user descriptor
      g0[1] = lds_addr;                             // lds_addr
      g0[2] = (unsigned)ga;                         // global_addr[31:0]
      g0[3] = (unsigned)((ga >> 32) & 0x01FFFFFFu) | (2u << 30); // addr[56:32], type=2
      v8i g1;
      g1[0] = 0x00010000;                           // data_size=1 (2B), no mask/pad
      g1[1] = 0;                                    // tensor_dim0[15:0]<<16 (td0=1<<30)
      g1[2] = 0x4000;                               // td0[31:16]=0x4000, td1 low=0
      g1[3] = 0x4000 | (64 << 16);                  // td1[31:16]=0x4000, tile_dim0=64
      g1[4] = 32;                                   // tile_dim1=32, tile_dim2=0
      g1[5] = DD;                                   // tensor_dim0_stride = 768
      g1[6] = 0;
      g1[7] = 0;
      v4i gz = (v4i)(0);
#if __clang_major__ >= 23
      v8i gz8 = (v8i)(0);
      __builtin_amdgcn_tensor_load_to_lds(g0, g1, gz, gz, gz8, 0);
#else
      __builtin_amdgcn_tensor_load_to_lds(g0, g1, gz, gz, 0);
#endif
      __builtin_amdgcn_s_wait_tensorcnt(0);
    }
#else
    { // fallback: lane loads one 64-element row of the V tile
      const bf16* vp = vv + ((size_t)b * NN + n + lane) * DD + h * HD;
#pragma unroll
      for (int j = 0; j < 4; ++j)
        *(v16bf*)&lv[lane * 64 + j * 16] = *(const v16bf*)(vp + j * 16);
    }
#endif

    // scores for two 16-key tiles
    const bf16* kp0 = kr + ((size_t)b * NN + n + col) * DD + h * HD + chalf;
    const bf16* kp1 = kr + ((size_t)b * NN + n + 16 + col) * DD + h * HD + chalf;
    v8f s0 = (v8f)(0.f), s1 = (v8f)(0.f);
    s0 = wmma_bf16(aq0, *(const v16bf*)kp0, s0);
    s0 = wmma_bf16(aq1, *(const v16bf*)(kp0 + 32), s0);
    s1 = wmma_bf16(aq0, *(const v16bf*)kp1, s1);
    s1 = wmma_bf16(aq1, *(const v16bf*)(kp1 + 32), s1);

#pragma unroll
    for (int r = 0; r < 8; ++r) {
      float a0 = s0[r] * 0.125f, a1 = s1[r] * 0.125f;
      float mx = fmaxf(a0, a1);
#pragma unroll
      for (int msk = 1; msk < 16; msk <<= 1) mx = fmaxf(mx, __shfl_xor(mx, msk, 32));
      float nm = fmaxf(m[r], mx);
      float corr = __expf(m[r] - nm);
      m[r] = nm;
      float p0 = __expf(a0 - nm), p1 = __expf(a1 - nm);
      float rs = p0 + p1;
#pragma unroll
      for (int msk = 1; msk < 16; msk <<= 1) rs += __shfl_xor(rs, msk, 32);
      l[r] = l[r] * corr + rs;
#pragma unroll
      for (int d = 0; d < 4; ++d) acc[d][r] *= corr;
      int rowc = r + ((lane >> 4) << 3);
      lp[rowc * 32 + col] = (bf16)p0;
      lp[rowc * 32 + 16 + col] = (bf16)p1;
    }
    // per-wave LDS ops are in-order; no block barrier needed (private tiles)

    // P (16x32) @ V (32x64)
    v16bf ap = *(const v16bf*)&lp[(lane & 15) * 32 + chalf];
    int noff = (lane >> 4) << 4;
#pragma unroll
    for (int d = 0; d < 4; ++d) {
      v16bf bm;
#pragma unroll
      for (int j = 0; j < 16; ++j) bm[j] = lv[(noff + j) * 64 + d * 16 + col];
      acc[d] = wmma_bf16(ap, bm, acc[d]);
    }
  }

  __syncthreads();  // all waves done before smem is repurposed for the merge

  float* lacc = (float*)smem + w * 1024;          // [16][64]
  float* lm = (float*)(smem + 32768);             // [8][16]
  float* ll = (float*)(smem + 33280);             // [8][16]
#pragma unroll
  for (int r = 0; r < 8; ++r) {
    int rowc = r + ((lane >> 4) << 3);
    if (col == 0) { lm[w * 16 + rowc] = m[r]; ll[w * 16 + rowc] = l[r]; }
#pragma unroll
    for (int d = 0; d < 4; ++d) lacc[rowc * 64 + d * 16 + col] = acc[d][r];
  }
  __syncthreads();

  for (int e = threadIdx.x; e < 1024; e += 256) {
    int s = e >> 6, d = e & 63;
    float M = -1e30f;
#pragma unroll
    for (int ww = 0; ww < 8; ++ww) M = fmaxf(M, lm[ww * 16 + s]);
    float num = 0.f, den = 0.f;
#pragma unroll
    for (int ww = 0; ww < 8; ++ww) {
      float f = __expf(lm[ww * 16 + s] - M);
      num += ((float*)smem)[ww * 1024 + s * 64 + d] * f;
      den += ll[ww * 16 + s] * f;
    }
    upd[((size_t)b * KK + s) * DD + h * HD + d] = num / den;
  }
}

// ---------------------------------------------------------------------------
// Kernel 7: GRU cell. one thread per (b,k,d)
// ---------------------------------------------------------------------------
__global__ void gru_kernel(const float* __restrict__ x, const float* __restrict__ hprev,
                           const float* __restrict__ wi, const float* __restrict__ wh,
                           const float* __restrict__ bi, const float* __restrict__ bh,
                           float* __restrict__ out) {
  int idx = blockIdx.x * 256 + threadIdx.x;
  if (idx >= BB * KK * DD) return;
  int r = idx / DD, d = idx - r * DD;
  const float* xr = x + (size_t)r * DD;
  const float* hr_ = hprev + (size_t)r * DD;
  float ir = bi[d], iz = bi[DD + d], inn = bi[2 * DD + d];
  float hrv = bh[d], hz = bh[DD + d], hn = bh[2 * DD + d];
  const float* wir = wi + (size_t)d * DD;
  const float* wiz = wi + (size_t)(DD + d) * DD;
  const float* win = wi + (size_t)(2 * DD + d) * DD;
  const float* whr = wh + (size_t)d * DD;
  const float* whz = wh + (size_t)(DD + d) * DD;
  const float* whn = wh + (size_t)(2 * DD + d) * DD;
  for (int c = 0; c < DD; ++c) {
    float xv = xr[c], hv = hr_[c];
    ir += xv * wir[c]; iz += xv * wiz[c]; inn += xv * win[c];
    hrv += hv * whr[c]; hz += hv * whz[c]; hn += hv * whn[c];
  }
  float rg = 1.f / (1.f + __expf(-(ir + hrv)));
  float zg = 1.f / (1.f + __expf(-(iz + hz)));
  float ng = tanhf(inn + rg * hn);
  out[idx] = (1.f - zg) * ng + zg * hprev[idx];
}

// ---------------------------------------------------------------------------
// Kernel 8: final blend + slot_pos + conf outputs
// ---------------------------------------------------------------------------
__global__ void final_kernel(const float* __restrict__ slots, const float* __restrict__ prev,
                             const float* __restrict__ confb, float* __restrict__ out) {
  int idx = blockIdx.x * 256 + threadIdx.x;
  const int NF = BB * KK * DD;
  const int NSP = BB * KK * 2;
  if (idx < NF) {
    float c = confb[idx / DD];
    out[idx] = slots[idx] * c + prev[idx] * (1.f - c);
  } else if (idx < NF + NSP) {
    int e = idx - NF;
    int k = (e >> 1) & 15;
    int comp = e & 1;
    out[idx] = (comp == 0) ? (float)(k >> 2) : (float)(k & 3);
  } else if (idx < NF + NSP + BB * KK) {
    out[idx] = confb[idx - NF - NSP];
  }
}

// ---------------------------------------------------------------------------
extern "C" void kernel_launch(void* const* d_in, const int* in_sizes, int n_in,
                              void* d_out, int out_size, void* d_ws, size_t ws_size,
                              hipStream_t stream) {
  const float* features = (const float*)d_in[0];
  const int*   fpos     = (const int*)d_in[1];
  const float* prev     = (const float*)d_in[2];
  const float* ln_in_w  = (const float*)d_in[3];
  const float* ln_in_b  = (const float*)d_in[4];
  const float* ln_sl_w  = (const float*)d_in[5];
  const float* ln_sl_b  = (const float*)d_in[6];
  const float* ln_ml_w  = (const float*)d_in[7];
  const float* ln_ml_b  = (const float*)d_in[8];
  const float* Wfp      = (const float*)d_in[9];
  const float* bfp      = (const float*)d_in[10];
  const float* wq       = (const float*)d_in[11];
  const float* bq       = (const float*)d_in[12];
  const float* wk       = (const float*)d_in[13];
  const float* bk       = (const float*)d_in[14];
  const float* wv       = (const float*)d_in[15];
  const float* bv       = (const float*)d_in[16];
  const float* wo       = (const float*)d_in[17];
  const float* bo       = (const float*)d_in[18];
  const float* gwi      = (const float*)d_in[19];
  const float* gwh      = (const float*)d_in[20];
  const float* gbi      = (const float*)d_in[21];
  const float* gbh      = (const float*)d_in[22];
  const float* mw1      = (const float*)d_in[23];
  const float* mb1      = (const float*)d_in[24];
  const float* mw2      = (const float*)d_in[25];
  const float* mb2      = (const float*)d_in[26];
  const float* cw       = (const float*)d_in[27];
  const float* cb       = (const float*)d_in[28];
  (void)in_sizes; (void)n_in; (void)out_size; (void)ws_size;

  char* W = (char*)d_ws;
  size_t off = 0;
  auto take = [&](size_t bytes) { size_t p = off; off += (bytes + 255) & ~(size_t)255; return p; };
  const size_t BN = (size_t)BB * NN;
  const size_t SLOT = (size_t)BB * KK * DD;

  bf16* krope  = (bf16*)(W + take(BN * DD * 2));
  bf16* vbuf   = (bf16*)(W + take(BN * DD * 2));
  bf16* wfp_bf = (bf16*)(W + take((size_t)DD * DF * 2));
  bf16* wk_bf  = (bf16*)(W + take((size_t)DD * DD * 2));
  bf16* wv_bf  = (bf16*)(W + take((size_t)DD * DD * 2));
  float* meanA = (float*)(W + take(BN * 4));
  float* rstdA = (float*)(W + take(BN * 4));
  float* sn    = (float*)(W + take(SLOT * 4));
  float* qraw  = (float*)(W + take(SLOT * 4));
  bf16* qbf    = (bf16*)(W + take(SLOT * 2));
  float* upd   = (float*)(W + take(SLOT * 4));
  float* upd2  = (float*)(W + take(SLOT * 4));
  float* bufA  = (float*)(W + take(SLOT * 4));
  float* bufB  = (float*)(W + take(SLOT * 4));
  float* bufM  = (float*)(W + take(SLOT * 4));
  float* snm   = (float*)(W + take(SLOT * 4));
  float* hbuf  = (float*)(W + take(SLOT * 4));
  float* confb = (float*)(W + take(BB * KK * 4));

  // one-time: weight conversion, LN stats, fused fp/k/v with RoPE
  cvt_bf16_kernel<<<(DD * DF + 255) / 256, 256, 0, stream>>>(Wfp, wfp_bf, DD * DF);
  cvt_bf16_kernel<<<(DD * DD + 255) / 256, 256, 0, stream>>>(wk, wk_bf, DD * DD);
  cvt_bf16_kernel<<<(DD * DD + 255) / 256, 256, 0, stream>>>(wv, wv_bf, DD * DD);
  stats_kernel<<<(int)(BN / 8), 256, 0, stream>>>(features, meanA, rstdA, (int)BN, DF);
  feat_kv_kernel<<<(int)(BN / 16), 256, 0, stream>>>(features, fpos, ln_in_w, ln_in_b,
                                                     wfp_bf, bfp, wk_bf, bk, wv_bf, bv,
                                                     krope, vbuf, meanA, rstdA);

  const int MV_GRID = (int)((SLOT + 255) / 256);
  const float* s_in[3]  = { prev, bufA, bufB };
  float*       s_out[3] = { bufA, bufB, bufA };

  for (int it = 0; it < 3; ++it) {
    const float* si = s_in[it];
    float* so = s_out[it];
    ln_rows_kernel<<<32, 256, 0, stream>>>(si, ln_sl_w, ln_sl_b, sn, BB * KK, DD);
    matvec_kernel<<<MV_GRID, 256, 0, stream>>>(sn, wq, bq, nullptr, qraw, BB * KK, DD, DD, 0);
    qrope_kernel<<<MV_GRID, 256, 0, stream>>>(qraw, qbf);
    attn_kernel<<<BB * HH, 256, 0, stream>>>(qbf, krope, vbuf, upd);
    matvec_kernel<<<MV_GRID, 256, 0, stream>>>(upd, wo, bo, nullptr, upd2, BB * KK, DD, DD, 0);
    gru_kernel<<<MV_GRID, 256, 0, stream>>>(upd2, si, gwi, gwh, gbi, gbh, bufM);
    ln_rows_kernel<<<32, 256, 0, stream>>>(bufM, ln_ml_w, ln_ml_b, snm, BB * KK, DD);
    matvec_kernel<<<MV_GRID, 256, 0, stream>>>(snm, mw1, mb1, nullptr, hbuf, BB * KK, DD, DD, 1);
    matvec_kernel<<<MV_GRID, 256, 0, stream>>>(hbuf, mw2, mb2, bufM, so, BB * KK, DD, DD, 0);
  }

  matvec_kernel<<<1, 256, 0, stream>>>(bufA, cw, cb, nullptr, confb, BB * KK, 1, DD, 2);
  const int TOT = BB * KK * DD + BB * KK * 2 + BB * KK;
  final_kernel<<<(TOT + 255) / 256, 256, 0, stream>>>(bufA, prev, confb, (float*)d_out);
}
